// MHA_18743237280339
// MI455X (gfx1250) — matrix operand order repaired and checked by reading, not compiled, not run on hardware
//
#include <hip/hip_runtime.h>
#include <hip/hip_bf16.h>
#include <math.h>

// ---------------------------------------------------------------------------
// MHA for MI455X (gfx1250): f16 WMMA (v_wmma_f32_16x16x32_f16) everywhere,
// flash-attention with online softmax, fp32 accumulation, and TDM
// (tensor_load_to_lds) double-buffered K/V staging shared by all 8 waves
// of a workgroup.
//   B=2, S=4096, D=512, H=8, hd=64.  scale = 1/sqrt(512) folded into Wq.
// ---------------------------------------------------------------------------

#define BATCH 2
#define SEQ   4096
#define DMODEL 512
#define NHEAD 8
#define HDIM  64
#define MTOT  (BATCH * SEQ)   // 8192

typedef _Float16 v16h __attribute__((ext_vector_type(16)));
typedef _Float16 v8h  __attribute__((ext_vector_type(8)));
typedef float    v8f  __attribute__((ext_vector_type(8)));
typedef unsigned int u32x4 __attribute__((ext_vector_type(4)));
typedef int          i32x8 __attribute__((ext_vector_type(8)));
typedef int          i32x4 __attribute__((ext_vector_type(4)));

#if defined(__has_builtin)
#if __has_builtin(__builtin_amdgcn_tensor_load_to_lds) && __has_builtin(__builtin_amdgcn_s_wait_tensorcnt)
#define USE_TDM 1
#endif
#endif
#ifndef USE_TDM
#define USE_TDM 0
#endif

__device__ __forceinline__ v8f wmma_f16(v16h a, v16h b, v8f c) {
    // emits v_wmma_f32_16x16x32_f16
    return __builtin_amdgcn_wmma_f32_16x16x32_f16(
        /*neg_a=*/false, a, /*neg_b=*/false, b,
        /*c_mod=*/(short)0, c, /*reuse_a=*/false, /*reuse_b=*/false);
}

__device__ __forceinline__ v8f zero8() {
    v8f z;
#pragma unroll
    for (int i = 0; i < 8; ++i) z[i] = 0.0f;
    return z;
}

__device__ __forceinline__ v16h cat8(v8h lo, v8h hi) {
    return __builtin_shufflevector(lo, hi, 0,1,2,3,4,5,6,7,8,9,10,11,12,13,14,15);
}

// A-fragment (16x32 f16, MxK), row-major source with leading dim `ld`.
// lane<16: K = kOff+0..15; lane>=16: kOff+16..31.  Two 16B loads.
__device__ __forceinline__ v16h load_a_frag(const _Float16* base, int row, int ld,
                                            int kOff, int g) {
    const _Float16* p = base + row * ld + kOff + 16 * g;
    v8h lo = *(const v8h*)p;
    v8h hi = *(const v8h*)(p + 8);
    return cat8(lo, hi);
}

// B-fragment (32x16 f16, KxN): contraction index contiguous in memory for a
// fixed column; p must already point at [col_base + kBase + 16*g].
__device__ __forceinline__ v16h load_b_frag(const _Float16* p) {
    v8h lo = *(const v8h*)p;
    v8h hi = *(const v8h*)(p + 8);
    return cat8(lo, hi);
}

#if USE_TDM
// ---------------------------------------------------------------------------
// TDM: DMA a 2-D f16 tile (tile_w x tile_h elements, row stride `row_stride`
// elements) from global memory to LDS.  D# per CDNA5 ISA ch.8:
//   group0: count=1 | lds_addr | global_addr[56:0] | type=2
//   group1: data_size=2B | tensor_dim0/1 | tile_dim0/1 | dim0 stride
//   groups 2/3 + trailing group: zero (<=2D tensor).  Tracked with TENSORcnt.
// ---------------------------------------------------------------------------
__device__ __forceinline__ void tdm_load_2d_f16(
    unsigned int lds_off, const void* gptr,
    unsigned int tile_w, unsigned int tile_h,
    unsigned int tensor_w, unsigned int tensor_h, unsigned int row_stride)
{
    unsigned long long ga = (unsigned long long)(uintptr_t)gptr;
    u32x4 g0;
    g0[0] = 1u;                                   // count=1 (valid user D#)
    g0[1] = lds_off;                              // LDS byte address
    g0[2] = (unsigned int)ga;                     // global_addr[31:0]
    g0[3] = ((unsigned int)(ga >> 32) & 0x01FFFFFFu) | (2u << 30);  // type=2
    i32x8 g1;
    g1[0] = (int)(1u << 16);                      // data_size = 1 -> 2 bytes
    g1[1] = (int)((tensor_w & 0xFFFFu) << 16);    // tensor_dim0[15:0] @ 63:48
    g1[2] = (int)((tensor_w >> 16) | ((tensor_h & 0xFFFFu) << 16));
    g1[3] = (int)((tensor_h >> 16) | (tile_w << 16));   // tile_dim0 @127:112
    g1[4] = (int)(tile_h & 0xFFFFu);              // tile_dim1 (tile_dim2=0)
    g1[5] = (int)row_stride;                      // tensor_dim0_stride[31:0]
    g1[6] = 0;
    g1[7] = 0;
    i32x4 gz4;
    gz4[0] = 0; gz4[1] = 0; gz4[2] = 0; gz4[3] = 0;
    i32x8 gz8;
#pragma unroll
    for (int i = 0; i < 8; ++i) gz8[i] = 0;
    // 6-arg form (clang-23 / therock-10.0 headers on this toolchain)
    __builtin_amdgcn_tensor_load_to_lds(g0, g1, gz4, gz4, gz8, 0);
}
#endif

// ---------------------------------------------------------------------------
// Kernel 1: fp32 -> f16 convert (with optional scale folding)
// ---------------------------------------------------------------------------
__global__ void mha_cvt_f16(const float* __restrict__ src, _Float16* __restrict__ dst,
                            int n, float scale) {
    int i = blockIdx.x * blockDim.x + threadIdx.x;
    if (i < n) dst[i] = (_Float16)(src[i] * scale);
}

// ---------------------------------------------------------------------------
// Kernel 2: fused QKV projection.
//   y[m,n] = sum_k x[m,k] * W[n,k] + b[n]     (W is [out,in])
//   z = blockIdx.z selects q/k/v.  Q,K stored [B,H,S,hd]; V stored [B,H,hd,S].
// ---------------------------------------------------------------------------
__global__ __launch_bounds__(256) void mha_qkv_proj(
    const _Float16* __restrict__ xh,
    const _Float16* __restrict__ wqh, const _Float16* __restrict__ wkh,
    const _Float16* __restrict__ wvh,
    const float* __restrict__ bq, const float* __restrict__ bk,
    const float* __restrict__ bv,
    _Float16* __restrict__ Qh, _Float16* __restrict__ Kh, _Float16* __restrict__ Vt,
    float qbias_scale)
{
    const int lane = threadIdx.x & 31;
    const int wave = threadIdx.x >> 5;
    const int col  = lane & 15;
    const int g    = lane >> 4;

    const int m0 = (blockIdx.x * 8 + wave) * 16;   // row tile base
    const int n0 = blockIdx.y * 16;                // col tile base
    const int which = blockIdx.z;                  // 0=q 1=k 2=v

    const _Float16* wh  = (which == 0) ? wqh : (which == 1) ? wkh : wvh;
    const float* bias   = (which == 0) ? bq  : (which == 1) ? bk  : bv;
    const float bscale  = (which == 0) ? qbias_scale : 1.0f;

    const int arow = m0 + col;
    const int bcol = n0 + col;

    v8f acc = zero8();
#pragma unroll 4
    for (int kt = 0; kt < DMODEL; kt += 32) {
        v16h a = load_a_frag(xh, arow, DMODEL, kt, g);
        v16h b = load_b_frag(wh + bcol * DMODEL + kt + 16 * g);
        acc = wmma_f16(a, b, acc);
    }

    const float bv_ = bias[bcol] * bscale;
    const int h = n0 >> 6;           // 16-wide tile never straddles a head
    const int d = (n0 & 63) + col;
    const int bidx = m0 >> 12;       // never straddles a batch

#pragma unroll
    for (int i = 0; i < 8; ++i) {
        const int m = m0 + i + 8 * g;
        const int s = m & (SEQ - 1);
        const _Float16 val = (_Float16)(acc[i] + bv_);
        const int headbase = (bidx * NHEAD + h);
        if (which == 0)      Qh[((size_t)headbase * SEQ + s) * HDIM + d] = val;
        else if (which == 1) Kh[((size_t)headbase * SEQ + s) * HDIM + d] = val;
        else                 Vt[((size_t)headbase * HDIM + d) * SEQ + s] = val;
    }
}

// ---------------------------------------------------------------------------
// Kernel 3: causal flash attention.
//   Block = 8 waves = 8 consecutive 16-query tiles of one (b,h).
//   Per 32-key step: K tile (32x64) and V tile (64x32, transposed layout) are
//   DMA'd into LDS by the TDM (wave 0 = producer, double-buffered, TENSORcnt),
//   then all 8 waves consume them: 4 WMMAs QK^T -> causal mask -> online
//   softmax (shfl_xor row reductions) -> P transposed via LDS -> 4 WMMAs P@V.
// ---------------------------------------------------------------------------
__global__ __launch_bounds__(256) void mha_flash_attn(
    const _Float16* __restrict__ Qh, const _Float16* __restrict__ Kh,
    const _Float16* __restrict__ Vt, _Float16* __restrict__ Ah)
{
    __shared__ __align__(16) _Float16 ktile[2][32 * HDIM];  // [k][d], 2x4KB
    __shared__ __align__(16) _Float16 vtile[2][HDIM * 32];  // [d][k], 2x4KB
    __shared__ __align__(16) _Float16 pbuf[8][16 * 32];     // per-wave P tile

    const int lane = threadIdx.x & 31;
    const int wave = threadIdx.x >> 5;
    const int col  = lane & 15;
    const int g    = lane >> 4;

    const int qTile = blockIdx.x * 8 + wave;
    const int h = blockIdx.y;
    const int b = blockIdx.z;
    const int qBase = qTile * 16;

    const size_t hb = (size_t)(b * NHEAD + h);
    const _Float16* Qp = Qh + hb * SEQ * HDIM;
    const _Float16* Kp = Kh + hb * SEQ * HDIM;
    const _Float16* Vp = Vt + hb * HDIM * SEQ;

    // Q A-fragments for d=0..31 and d=32..63 (kept resident)
    const v16h aq0 = load_a_frag(Qp, qBase + col, HDIM, 0,  g);
    const v16h aq1 = load_a_frag(Qp, qBase + col, HDIM, 32, g);

    v8f o0 = zero8(), o1 = zero8(), o2 = zero8(), o3 = zero8();
    float m8[8], l8[8];
#pragma unroll
    for (int i = 0; i < 8; ++i) { m8[i] = -3.0e38f; l8[i] = 0.0f; }

    _Float16* pb = &pbuf[wave][0];

    // number of 32-key steps covering the causal range of the LAST wave
    const int niter = blockIdx.x * 4 + 4;

#if USE_TDM
    if (wave == 0) {
        // prime the pipeline: step 0 into buffer 0
        tdm_load_2d_f16((unsigned int)(uintptr_t)&ktile[0][0], Kp,
                        HDIM, 32, HDIM, SEQ, HDIM);
        tdm_load_2d_f16((unsigned int)(uintptr_t)&vtile[0][0], Vp,
                        32, HDIM, SEQ, HDIM, SEQ);
    }
#endif

    for (int iter = 0; iter < niter; ++iter) {
        const int kBase = iter * 32;
        const int bs = iter & 1;

        __syncthreads();   // everyone done reading buffer (iter+1)&1
#if USE_TDM
        if (wave == 0) {
            if (iter + 1 < niter) {
                const int kn = kBase + 32;
                const int ns = (iter + 1) & 1;
                tdm_load_2d_f16((unsigned int)(uintptr_t)&ktile[ns][0],
                                Kp + (size_t)kn * HDIM,
                                HDIM, 32, HDIM, (unsigned)(SEQ - kn), HDIM);
                tdm_load_2d_f16((unsigned int)(uintptr_t)&vtile[ns][0],
                                Vp + kn,
                                32, HDIM, (unsigned)(SEQ - kn), HDIM, SEQ);
                __builtin_amdgcn_s_wait_tensorcnt(2);  // current step done
            } else {
                __builtin_amdgcn_s_wait_tensorcnt(0);
            }
        }
#endif
        __syncthreads();   // buffer `bs` ready for all waves

        if (kBase >= qBase + 16) continue;  // past causal limit: barrier-only

        // ---- scores: two 16x16 C tiles over 32 keys ----------------------
        v8f s0 = zero8(), s1 = zero8();
        {
#if USE_TDM
            const _Float16* kb = &ktile[bs][0];
            v16h bk;
            bk = load_b_frag(kb + (col)      * HDIM +  0 + 16 * g);
            s0 = wmma_f16(aq0, bk, s0);
            bk = load_b_frag(kb + (16 + col) * HDIM +  0 + 16 * g);
            s1 = wmma_f16(aq0, bk, s1);
            bk = load_b_frag(kb + (col)      * HDIM + 32 + 16 * g);
            s0 = wmma_f16(aq1, bk, s0);
            bk = load_b_frag(kb + (16 + col) * HDIM + 32 + 16 * g);
            s1 = wmma_f16(aq1, bk, s1);
#else
            v16h bk;
            bk = load_b_frag(Kp + (kBase + col)      * HDIM +  0 + 16 * g);
            s0 = wmma_f16(aq0, bk, s0);
            bk = load_b_frag(Kp + (kBase + 16 + col) * HDIM +  0 + 16 * g);
            s1 = wmma_f16(aq0, bk, s1);
            bk = load_b_frag(Kp + (kBase + col)      * HDIM + 32 + 16 * g);
            s0 = wmma_f16(aq1, bk, s0);
            bk = load_b_frag(Kp + (kBase + 16 + col) * HDIM + 32 + 16 * g);
            s1 = wmma_f16(aq1, bk, s1);
#endif
        }

        // ---- causal mask + online softmax (rows live in 16-lane halves) --
#pragma unroll
        for (int i = 0; i < 8; ++i) {
            const int q = qBase + i + 8 * g;
            float v0 = (kBase + col      <= q) ? s0[i] : -3.0e38f;
            float v1 = (kBase + 16 + col <= q) ? s1[i] : -3.0e38f;

            float rm = fmaxf(v0, v1);
            rm = fmaxf(rm, __shfl_xor(rm, 1));
            rm = fmaxf(rm, __shfl_xor(rm, 2));
            rm = fmaxf(rm, __shfl_xor(rm, 4));
            rm = fmaxf(rm, __shfl_xor(rm, 8));

            const float nm = fmaxf(m8[i], rm);
            const float p0 = __expf(v0 - nm);
            const float p1 = __expf(v1 - nm);

            float rs = p0 + p1;
            rs += __shfl_xor(rs, 1);
            rs += __shfl_xor(rs, 2);
            rs += __shfl_xor(rs, 4);
            rs += __shfl_xor(rs, 8);

            const float corr = __expf(m8[i] - nm);
            l8[i] = l8[i] * corr + rs;
            m8[i] = nm;
            o0[i] *= corr; o1[i] *= corr; o2[i] *= corr; o3[i] *= corr;

            const int prow = i + 8 * g;
            pb[prow * 32 + col]      = (_Float16)p0;
            pb[prow * 32 + 16 + col] = (_Float16)p1;
        }

        // ---- P: C-layout -> A-layout via LDS (same-wave DS ops in order) --
        v8h plo = *(const v8h*)(pb + col * 32 + 16 * g);
        v8h phi = *(const v8h*)(pb + col * 32 + 8 + 16 * g);
        v16h pa = cat8(plo, phi);

        // ---- O += P @ V ---------------------------------------------------
#if USE_TDM
        const _Float16* vb = &vtile[bs][0];
        o0 = wmma_f16(pa, load_b_frag(vb + ( 0 + col) * 32 + 16 * g), o0);
        o1 = wmma_f16(pa, load_b_frag(vb + (16 + col) * 32 + 16 * g), o1);
        o2 = wmma_f16(pa, load_b_frag(vb + (32 + col) * 32 + 16 * g), o2);
        o3 = wmma_f16(pa, load_b_frag(vb + (48 + col) * 32 + 16 * g), o3);
#else
        o0 = wmma_f16(pa, load_b_frag(Vp + ( 0 + col) * SEQ + kBase + 16 * g), o0);
        o1 = wmma_f16(pa, load_b_frag(Vp + (16 + col) * SEQ + kBase + 16 * g), o1);
        o2 = wmma_f16(pa, load_b_frag(Vp + (32 + col) * SEQ + kBase + 16 * g), o2);
        o3 = wmma_f16(pa, load_b_frag(Vp + (48 + col) * SEQ + kBase + 16 * g), o3);
#endif
    }

    // ---- epilogue: normalize rows, write f16 [B,S,D] attention output ----
#pragma unroll
    for (int i = 0; i < 8; ++i) {
        const float inv = 1.0f / l8[i];
        const size_t row = (size_t)(b * SEQ + qBase + i + 8 * g) * DMODEL + h * HDIM + col;
        Ah[row]      = (_Float16)(o0[i] * inv);
        Ah[row + 16] = (_Float16)(o1[i] * inv);
        Ah[row + 32] = (_Float16)(o2[i] * inv);
        Ah[row + 48] = (_Float16)(o3[i] * inv);
    }
}

// ---------------------------------------------------------------------------
// Kernel 4: output projection, fp32 result.
// ---------------------------------------------------------------------------
__global__ __launch_bounds__(256) void mha_out_proj(
    const _Float16* __restrict__ Ah, const _Float16* __restrict__ woh,
    const float* __restrict__ bo, float* __restrict__ out)
{
    const int lane = threadIdx.x & 31;
    const int wave = threadIdx.x >> 5;
    const int col  = lane & 15;
    const int g    = lane >> 4;

    const int m0 = (blockIdx.x * 8 + wave) * 16;
    const int n0 = blockIdx.y * 16;
    const int arow = m0 + col;
    const int bcol = n0 + col;

    v8f acc = zero8();
#pragma unroll 4
    for (int kt = 0; kt < DMODEL; kt += 32) {
        v16h a = load_a_frag(Ah, arow, DMODEL, kt, g);
        v16h b = load_b_frag(woh + bcol * DMODEL + kt + 16 * g);
        acc = wmma_f16(a, b, acc);
    }

    const float bv_ = bo[bcol];
#pragma unroll
    for (int i = 0; i < 8; ++i) {
        const int m = m0 + i + 8 * g;
        out[(size_t)m * DMODEL + bcol] = acc[i] + bv_;
    }
}

// ---------------------------------------------------------------------------
// Host-side launcher
// ---------------------------------------------------------------------------
extern "C" void kernel_launch(void* const* d_in, const int* in_sizes, int n_in,
                              void* d_out, int out_size, void* d_ws, size_t ws_size,
                              hipStream_t stream) {
    const float* x    = (const float*)d_in[0];
    const float* wq_w = (const float*)d_in[1];
    const float* wq_b = (const float*)d_in[2];
    const float* wk_w = (const float*)d_in[3];
    const float* wk_b = (const float*)d_in[4];
    const float* wv_w = (const float*)d_in[5];
    const float* wv_b = (const float*)d_in[6];
    const float* wo_w = (const float*)d_in[7];
    const float* wo_b = (const float*)d_in[8];
    float* out = (float*)d_out;

    // workspace layout (all f16)
    char* ws = (char*)d_ws;
    const size_t SZ_X = (size_t)MTOT * DMODEL * sizeof(_Float16);       // 8 MiB
    const size_t SZ_W = (size_t)DMODEL * DMODEL * sizeof(_Float16);     // 0.5 MiB
    _Float16* xh  = (_Float16*)(ws);
    _Float16* wqh = (_Float16*)(ws + SZ_X);
    _Float16* wkh = (_Float16*)(ws + SZ_X + SZ_W);
    _Float16* wvh = (_Float16*)(ws + SZ_X + 2 * SZ_W);
    _Float16* woh = (_Float16*)(ws + SZ_X + 3 * SZ_W);
    _Float16* Qh  = (_Float16*)(ws + SZ_X + 4 * SZ_W);
    _Float16* Kh  = (_Float16*)(ws + 2 * SZ_X + 4 * SZ_W);
    _Float16* Vt  = (_Float16*)(ws + 3 * SZ_X + 4 * SZ_W);
    _Float16* Ah  = (_Float16*)(ws + 4 * SZ_X + 4 * SZ_W);

    const float inv_sqrt_d = 1.0f / sqrtf((float)DMODEL);  // folded into Wq/bq

    const int nX = MTOT * DMODEL;
    const int nW = DMODEL * DMODEL;
    mha_cvt_f16<<<(nX + 255) / 256, 256, 0, stream>>>(x,    xh,  nX, 1.0f);
    mha_cvt_f16<<<(nW + 255) / 256, 256, 0, stream>>>(wq_w, wqh, nW, inv_sqrt_d);
    mha_cvt_f16<<<(nW + 255) / 256, 256, 0, stream>>>(wk_w, wkh, nW, 1.0f);
    mha_cvt_f16<<<(nW + 255) / 256, 256, 0, stream>>>(wv_w, wvh, nW, 1.0f);
    mha_cvt_f16<<<(nW + 255) / 256, 256, 0, stream>>>(wo_w, woh, nW, 1.0f);

    // QKV projection: 512 m-tiles (8 per block) x 32 n-tiles x {q,k,v}
    mha_qkv_proj<<<dim3(MTOT / 128, DMODEL / 16, 3), 256, 0, stream>>>(
        xh, wqh, wkh, wvh, wq_b, wk_b, wv_b, Qh, Kh, Vt, inv_sqrt_d);

    // Flash attention: 256 q-tiles (8 per block) x H x B
    mha_flash_attn<<<dim3(SEQ / 16 / 8, NHEAD, BATCH), 256, 0, stream>>>(
        Qh, Kh, Vt, Ah);

    // Output projection (fp32 result)
    mha_out_proj<<<dim3(MTOT / 128, DMODEL / 16), 256, 0, stream>>>(
        Ah, woh, wo_b, out);
}
